// GAT_18880676233585
// MI455X (gfx1250) — matrix operand order repaired and checked by reading, not compiled, not run on hardware
//
#include <hip/hip_runtime.h>
#include <hip/hip_bf16.h>

// ---------------------------------------------------------------------------
// GAT (3 layers, 2 heads, d=64) + mean pool + MLP for MI455X (gfx1250).
// Dense projections use v_wmma_f32_16x16x32_bf16 (bf16 A/B, f32 accum).
// Weights are pre-packed into the exact CDNA5 B-fragment register image so
// each B tile is two global_load_b128 per lane; A fragments load as 4x b128.
// Edge softmax/aggregation uses f32 global atomics; z/out (51MB each) are
// L2-resident (192MB L2), so gather/scatter is L2-bandwidth bound, not HBM.
// ---------------------------------------------------------------------------

#define NN 100000            // nodes
#define NE 1600000           // edges
#define E2 (NE + NN)         // edges incl. self loops
#define NG 512               // graphs
#define CAT 128              // H * d = 2 * 64

typedef __attribute__((ext_vector_type(16))) __bf16 v16bf;
typedef __attribute__((ext_vector_type(8)))  float  v8f;

union BF16x16 { v16bf v; unsigned short u[16]; uint4 q[2]; };

__device__ __forceinline__ unsigned short f2bf(float f) {
    unsigned u = __float_as_uint(f);
    u += 0x7FFFu + ((u >> 16) & 1u);       // round-to-nearest-even
    return (unsigned short)(u >> 16);
}

// monotone float <-> uint key for atomicMax-based segment max
__device__ __forceinline__ unsigned fkey(float f) {
    unsigned b = __float_as_uint(f);
    return (b & 0x80000000u) ? ~b : (b | 0x80000000u);
}
__device__ __forceinline__ float finv(unsigned k) {
    unsigned b = (k & 0x80000000u) ? (k ^ 0x80000000u) : ~k;
    return __uint_as_float(b);
}

__device__ __forceinline__ void edge_pair(int i, const int* __restrict__ ed,
                                          const int* __restrict__ er, int& s, int& t) {
    if (i < NE) { s = ed[i]; t = er[i]; } else { s = i - NE; t = i - NE; }
}

// --- weights [H, F, 64] f32 -> bf16 packed as WMMA B-fragment register image.
// wpack[(((ks*8 + c)*32 + lane)*16 + e)] holds B[k = ks*32 + (e&7) + 16*(e>=8) + 8*(lane>=16)]
//                                              [col = c*16 + (lane&15)]
// with B[f, h*64+d] = W[h, f, d].  Each lane's 16 bf16 = 32 contiguous bytes.
__global__ void k_wpack(const float* __restrict__ W, unsigned short* __restrict__ wpack, int F) {
    int idx = blockIdx.x * blockDim.x + threadIdx.x;
    if (idx >= F * CAT) return;
    int e    = idx & 15;
    int lane = (idx >> 4) & 31;
    int tile = idx >> 9;
    int c    = tile & 7;
    int ks   = tile >> 3;
    int k    = ks * 32 + (e & 7) + ((e & 8) ? 16 : 0) + ((lane & 16) ? 8 : 0);     // CDNA5 16-bit B layout
    int col  = c * 16 + (lane & 15);
    int h    = col >> 6, d = col & 63;
    wpack[idx] = f2bf(W[((size_t)h * F + k) * 64 + d]);
}

// --- WMMA projection: z[n, h*64+d] = sum_f in[n,f] * Wcat[f, h*64+d] --------
// one wave per 16-row stripe; 8 column tiles of 16; K-steps of 32 in bf16.
template <int F>
__global__ __launch_bounds__(32) void k_proj(const float* __restrict__ in,
                                             const unsigned short* __restrict__ wpack,
                                             float* __restrict__ z) {
    constexpr int KS = F / 32;
    const int lane = threadIdx.x;
    const int n0   = blockIdx.x * 16;
    const int row  = n0 + (lane & 15);
    const int khiA = (lane & 16) >> 1;   // +8 K offset for upper half-wave (A frag)

    // A fragments: 16x32 bf16 per K-step; lane holds row M = lane&15.
    // Lane's elements are two contiguous 8-float runs: [khiA, khiA+8) and
    // [16+khiA, 24+khiA)  (k = e + (e&8) + khiA)  -> 4x float4 loads.
    BF16x16 A[KS];
#pragma unroll
    for (int ks = 0; ks < KS; ++ks) {
        const float* src = in + (size_t)row * F + ks * 32 + khiA;
        float4 f0 = *(const float4*)(src);
        float4 f1 = *(const float4*)(src + 4);
        float4 f2 = *(const float4*)(src + 16);
        float4 f3 = *(const float4*)(src + 20);
        A[ks].u[0]  = f2bf(f0.x); A[ks].u[1]  = f2bf(f0.y);
        A[ks].u[2]  = f2bf(f0.z); A[ks].u[3]  = f2bf(f0.w);
        A[ks].u[4]  = f2bf(f1.x); A[ks].u[5]  = f2bf(f1.y);
        A[ks].u[6]  = f2bf(f1.z); A[ks].u[7]  = f2bf(f1.w);
        A[ks].u[8]  = f2bf(f2.x); A[ks].u[9]  = f2bf(f2.y);
        A[ks].u[10] = f2bf(f2.z); A[ks].u[11] = f2bf(f2.w);
        A[ks].u[12] = f2bf(f3.x); A[ks].u[13] = f2bf(f3.y);
        A[ks].u[14] = f2bf(f3.z); A[ks].u[15] = f2bf(f3.w);
    }

#pragma unroll
    for (int c = 0; c < 8; ++c) {
        v8f acc = {0.f, 0.f, 0.f, 0.f, 0.f, 0.f, 0.f, 0.f};
#pragma unroll
        for (int ks = 0; ks < KS; ++ks) {
            BF16x16 B;
            const uint4* p = (const uint4*)(wpack + (size_t)(((ks * 8 + c) * 32 + lane) * 16));
            B.q[0] = p[0];
            B.q[1] = p[1];
            acc = __builtin_amdgcn_wmma_f32_16x16x32_bf16(
                false, A[ks].v, false, B.v, (short)0, acc, false, false);
        }
        const int col  = c * 16 + (lane & 15);
        const int mofs = (lane & 16) >> 1;               // C/D layout: M = r + 8*hi
#pragma unroll
        for (int r = 0; r < 8; ++r)
            z[(size_t)(n0 + r + mofs) * CAT + col] = acc[r];
    }
}

// --- per-node attention scalars: sd[n,h] = z_row . a[h,:64], sr = . a[h,64:]
__global__ void k_scores(const float* __restrict__ z, const float* __restrict__ a,
                         float* __restrict__ sd, float* __restrict__ sr) {
    int idx = blockIdx.x * blockDim.x + threadIdx.x;
    if (idx >= NN * 2) return;
    int n = idx >> 1, h = idx & 1;
    const float* zr = z + (size_t)n * CAT + h * 64;
    const float* ad = a + h * 128;
    const float* ar = ad + 64;
    float s0 = 0.f, s1 = 0.f;
#pragma unroll 8
    for (int d = 0; d < 64; ++d) { float v = zr[d]; s0 += v * ad[d]; s1 += v * ar[d]; }
    sd[idx] = s0; sr[idx] = s1;
}

__device__ __forceinline__ float edge_score(int s, int t, int h,
                                            const float* __restrict__ sd,
                                            const float* __restrict__ sr) {
    float sc = sd[s * 2 + h] + sr[t * 2 + h];
    return sc > 0.f ? sc : 0.01f * sc;                   // leaky_relu(0.01)
}

// --- pass 1: segment max via atomicMax on monotone key -----------------------
__global__ void k_edge_max(const int* __restrict__ ed, const int* __restrict__ er,
                           const float* __restrict__ sd, const float* __restrict__ sr,
                           unsigned* __restrict__ mkey) {
    int idx = blockIdx.x * blockDim.x + threadIdx.x;
    if (idx >= E2 * 2) return;
    int i = idx >> 1, h = idx & 1, s, t;
    edge_pair(i, ed, er, s, t);
    atomicMax(&mkey[t * 2 + h], fkey(edge_score(s, t, h, sd, sr)));
}

// --- pass 2: segment sum of exp(e - m) ---------------------------------------
__global__ void k_edge_sum(const int* __restrict__ ed, const int* __restrict__ er,
                           const float* __restrict__ sd, const float* __restrict__ sr,
                           const unsigned* __restrict__ mkey, float* __restrict__ den) {
    int idx = blockIdx.x * blockDim.x + threadIdx.x;
    if (idx >= E2 * 2) return;
    int i = idx >> 1, h = idx & 1, s, t;
    edge_pair(i, ed, er, s, t);
    float m = finv(mkey[t * 2 + h]);
    atomicAdd(&den[t * 2 + h], __expf(edge_score(s, t, h, sd, sr) - m));
}

// --- pass 3: out[er] += alpha * z[ed]; one wave per (edge, head) -------------
__global__ __launch_bounds__(256) void k_edge_msg(
        const int* __restrict__ ed, const int* __restrict__ er,
        const float* __restrict__ sd, const float* __restrict__ sr,
        const unsigned* __restrict__ mkey, const float* __restrict__ den,
        const float* __restrict__ z, float* __restrict__ out) {
    int gw   = (blockIdx.x * blockDim.x + threadIdx.x) >> 5;
    int lane = threadIdx.x & 31;
    if (gw >= E2 * 2) return;
    int i = gw >> 1, h = gw & 1, s, t;
    edge_pair(i, ed, er, s, t);
    float m     = finv(mkey[t * 2 + h]);
    float alpha = __expf(edge_score(s, t, h, sd, sr) - m) / den[t * 2 + h];
    const float2 zz = *(const float2*)(z + (size_t)s * CAT + h * 64 + lane * 2);
    float* op = out + (size_t)t * CAT + h * 64 + lane * 2;
    atomicAdd(op,     zz.x * alpha);
    atomicAdd(op + 1, zz.y * alpha);
}

// --- SELU --------------------------------------------------------------------
__global__ void k_selu(float* __restrict__ x, int nel) {
    int i = blockIdx.x * blockDim.x + threadIdx.x;
    if (i >= nel) return;
    float v = x[i];
    x[i] = 1.0507009873554805f * (v > 0.f ? v : 1.6732632423543772f * (__expf(v) - 1.f));
}

// --- graph mean pool (accumulate) -------------------------------------------
__global__ void k_pool(const float* __restrict__ h, const int* __restrict__ gid,
                       float* __restrict__ gsum, float* __restrict__ cnt) {
    int idx = blockIdx.x * blockDim.x + threadIdx.x;
    if (idx >= NN * CAT) return;
    int n = idx >> 7, c = idx & 127;
    int g = gid[n];
    atomicAdd(&gsum[(size_t)g * CAT + c], h[idx]);
    if (c == 0) atomicAdd(&cnt[g], 1.f);
}

// --- MLP head: one wave per graph -------------------------------------------
__global__ __launch_bounds__(32) void k_mlp(const float* __restrict__ gsum,
                                            const float* __restrict__ cnt,
                                            const float* __restrict__ w1,
                                            const float* __restrict__ b1,
                                            const float* __restrict__ w2,
                                            const float* __restrict__ b2,
                                            float* __restrict__ outp) {
    int g = blockIdx.x, lane = threadIdx.x;
    float c   = cnt[g];
    float inv = c > 0.f ? 1.f / c : 0.f;
    float a0 = b1[lane], a1 = b1[lane + 32];
    for (int k = 0; k < CAT; ++k) {
        float ge = gsum[(size_t)g * CAT + k] * inv;
        a0 += ge * w1[k * 64 + lane];
        a1 += ge * w1[k * 64 + lane + 32];
    }
    a0 = a0 > 0.f ? a0 : 0.f;
    a1 = a1 > 0.f ? a1 : 0.f;
    float p = a0 * w2[lane] + a1 * w2[lane + 32];
    for (int off = 16; off > 0; off >>= 1) p += __shfl_down(p, off, 32);
    if (lane == 0) outp[g] = p + b2[0];
}

// ---------------------------------------------------------------------------
extern "C" void kernel_launch(void* const* d_in, const int* in_sizes, int n_in,
                              void* d_out, int out_size, void* d_ws, size_t ws_size,
                              hipStream_t stream) {
    (void)in_sizes; (void)n_in; (void)out_size; (void)ws_size;
    const float* nf  = (const float*)d_in[0];
    const int*   ed  = (const int*)d_in[2];
    const int*   er  = (const int*)d_in[3];
    const int*   gid = (const int*)d_in[4];
    const float* Wl[3] = {(const float*)d_in[5], (const float*)d_in[7], (const float*)d_in[9]};
    const float* al[3] = {(const float*)d_in[6], (const float*)d_in[8], (const float*)d_in[10]};
    const float* mw1 = (const float*)d_in[11];
    const float* mb1 = (const float*)d_in[12];
    const float* mw2 = (const float*)d_in[13];
    const float* mb2 = (const float*)d_in[14];
    float* out = (float*)d_out;

    // workspace carve-out
    char*  ws  = (char*)d_ws;
    size_t off = 0;
    auto carve = [&](size_t bytes) -> void* {
        void* p = ws + off; off += (bytes + 255) & ~(size_t)255; return p;
    };
    float*          z     = (float*)carve((size_t)NN * CAT * 4);
    float*          hA    = (float*)carve((size_t)NN * CAT * 4);
    float*          hB    = (float*)carve((size_t)NN * CAT * 4);
    float*          sd    = (float*)carve((size_t)NN * 2 * 4);
    float*          sr    = (float*)carve((size_t)NN * 2 * 4);
    unsigned*       mkey  = (unsigned*)carve((size_t)NN * 2 * 4);
    float*          den   = (float*)carve((size_t)NN * 2 * 4);
    unsigned short* wpack = (unsigned short*)carve((size_t)128 * CAT * 2);
    float*          gsum  = (float*)carve((size_t)NG * CAT * 4);
    float*          cnt   = (float*)carve((size_t)NG * 4);

    const float* lin[3]  = {nf, hA, hB};
    float*       lout[3] = {hA, hB, hA};
    const int    Fs[3]   = {64, 128, 128};

    const int eb = 256;
    const int eg_scalar = (E2 * 2 + eb - 1) / eb;                 // thread per (edge, head)
    const long long msg_threads = (long long)E2 * 2 * 32;         // wave per (edge, head)
    const int eg_msg = (int)((msg_threads + eb - 1) / eb);

    for (int l = 0; l < 3; ++l) {
        const int F = Fs[l];
        k_wpack<<<(F * CAT + 255) / 256, 256, 0, stream>>>(Wl[l], wpack, F);
        if (F == 64)
            k_proj<64><<<NN / 16, 32, 0, stream>>>(lin[l], wpack, z);
        else
            k_proj<128><<<NN / 16, 32, 0, stream>>>(lin[l], wpack, z);
        k_scores<<<(NN * 2 + 255) / 256, 256, 0, stream>>>(z, al[l], sd, sr);

        hipMemsetAsync(mkey, 0, (size_t)NN * 2 * 4, stream);
        hipMemsetAsync(den,  0, (size_t)NN * 2 * 4, stream);
        hipMemsetAsync(lout[l], 0, (size_t)NN * CAT * 4, stream);

        k_edge_max<<<eg_scalar, eb, 0, stream>>>(ed, er, sd, sr, mkey);
        k_edge_sum<<<eg_scalar, eb, 0, stream>>>(ed, er, sd, sr, mkey, den);
        k_edge_msg<<<eg_msg, eb, 0, stream>>>(ed, er, sd, sr, mkey, den, z, lout[l]);

        if (l < 2)
            k_selu<<<(NN * CAT + 255) / 256, 256, 0, stream>>>(lout[l], NN * CAT);
    }

    hipMemsetAsync(gsum, 0, (size_t)NG * CAT * 4, stream);
    hipMemsetAsync(cnt,  0, (size_t)NG * 4, stream);
    k_pool<<<(NN * CAT + 255) / 256, 256, 0, stream>>>(lout[2], gid, gsum, cnt);
    k_mlp<<<NG, 32, 0, stream>>>(gsum, cnt, mw1, mb1, mw2, mb2, out);
}
